// RWKVEncoder_86303072846164
// MI455X (gfx1250) — compile-verified, matrix operand order checked by reading, not run on hardware
//
#include <hip/hip_runtime.h>

#define TID (threadIdx.x)

constexpr int kL = 12, kD = 768, kB = 4, kT = 256, kF = 3072;
constexpr int kDD = kD * kD;          // 589824
constexpr int kDF = kD * kF;          // 2359296
constexpr int kKchD = kD / 32;        // 24 K-chunks for K=768
constexpr int kKchF = kF / 32;        // 96 K-chunks for K=3072
constexpr int kNtD = kD / 16;         // 48 N-tiles for N=768
constexpr int kNtF = kF / 16;         // 192 N-tiles for N=3072
constexpr int kPkD = kKchD * 512;     // 12288 packed-A elems for K=768
constexpr int kPkF = kKchF * 512;     // 49152 packed-A elems for K=3072
constexpr int kNWG = 24;
constexpr int kThreads = 256;
constexpr int kWavesPerWG = kThreads / 32;
constexpr float kNegInf = -1e30f;

typedef __attribute__((ext_vector_type(16))) __bf16 v16bf;
typedef __attribute__((ext_vector_type(8)))  float  v8f;
typedef __attribute__((ext_vector_type(4)))  int    v4i;

#if defined(__has_builtin)
#if __has_builtin(__builtin_amdgcn_global_load_async_to_lds_b128)
#define ASYNC_LDS_COPY 1
#endif
#endif

// ---------------- fp32 -> bf16 (round to nearest even) ----------------
__device__ __forceinline__ unsigned short f2bf_u(float f) {
  union { float f; unsigned u; } in; in.f = f;
  unsigned u = in.u + 0x7FFFu + ((in.u >> 16) & 1u);
  return (unsigned short)(u >> 16);
}

// ---------------- weight pack: [K,N] fp32 -> WMMA B-fragment bf16 ------
// CDNA5 16-bit B operand (32x16): lane n = lane&15; lanes 0-15 hold
// K = kc*32 + {0..7, 16..23}; lanes 16-31 hold K = kc*32 + {8..15, 24..31}.
__global__ void pack_weights(const float* __restrict__ src,
                             unsigned short* __restrict__ dst,
                             int K, int N, int total) {
  int gid = blockIdx.x * blockDim.x + TID;
  if (gid >= total) return;
  int kch = K >> 5, ntiles = N >> 4;
  int per = ntiles * kch * 32;
  int l = gid / per, r = gid % per;
  int nt = r / (kch * 32);
  int r2 = r % (kch * 32);
  int kc = r2 >> 5, lane = r2 & 31;
  const float* s = src + (size_t)l * K * N;
  unsigned short* d = dst + (size_t)l * K * N +
                      (((size_t)nt * kch + kc) * 32 + lane) * 16;
  int n  = nt * 16 + (lane & 15);
  int kb = kc * 32 + ((lane >> 4) ? 8 : 0);
#pragma unroll
  for (int i = 0; i < 8; ++i) {
    d[i]     = f2bf_u(s[(size_t)(kb + i) * N + n]);
    d[8 + i] = f2bf_u(s[(size_t)(kb + 16 + i) * N + n]);
  }
}

// ---------------- embedding + ln0 -------------------------------------
__global__ void embed_ln0(const int* __restrict__ tokens,
                          const float* __restrict__ emb,
                          const float* __restrict__ w,
                          const float* __restrict__ b,
                          float* __restrict__ x_seq) {
  __shared__ float r1[kThreads], r2s[kThreads];
  int bt = blockIdx.x;                       // b*T + t
  int tok = tokens[bt];
  const float* e = emb + (size_t)tok * kD;
  float s = 0.f, s2 = 0.f;
  for (int d = TID; d < kD; d += kThreads) { float v = e[d]; s += v; s2 += v * v; }
  r1[TID] = s; r2s[TID] = s2; __syncthreads();
  for (int o = kThreads / 2; o > 0; o >>= 1) {
    if (TID < o) { r1[TID] += r1[TID + o]; r2s[TID] += r2s[TID + o]; }
    __syncthreads();
  }
  float mu = r1[0] * (1.f / kD);
  float var = r2s[0] * (1.f / kD) - mu * mu;
  float rstd = rsqrtf(var + 1e-5f);
  for (int d = TID; d < kD; d += kThreads)
    x_seq[(size_t)bt * kD + d] = (e[d] - mu) * rstd * w[d] + b[d];
}

// ---------------- state init ------------------------------------------
__global__ void init_state(float* xa, float* xf, float* aa, float* bb, float* pp,
                           unsigned* bar) {
  int i = blockIdx.x * blockDim.x + TID;
  int n = kL * kB * kD;
  if (i < n) { xa[i] = 0.f; xf[i] = 0.f; aa[i] = 0.f; bb[i] = 0.f; pp[i] = kNegInf; }
  if (i == 0) { bar[0] = 0u; bar[1] = 0u; }
}

// ---------------- persistent kernel -----------------------------------
struct RwkvParams {
  const unsigned short *Wk, *Wv, *Wr, *Wo, *Fr, *Fk, *Fv;   // packed bf16 B
  const float *ln1_w, *ln1_b, *ln2_w, *ln2_b;
  const float *mk, *mv, *mr, *tf, *td, *fmk, *fmr;
  const float *lnow, *lnob;
  const float *x_seq;
  float *x, *xa, *xf, *aa, *bb, *pp;
  float *xk, *xv, *xr, *kout, *vout, *rout, *rg;
  float *xk2, *xr2, *r2raw, *kkraw, *kk, *r2sig;
  unsigned short *pk_xk, *pk_xv, *pk_xr, *pk_rg, *pk_xk2, *pk_xr2, *pk_kk;
  float *out;
  unsigned *bar;
};

__device__ void grid_barrier(unsigned* cnt, unsigned* gen) {
  __syncthreads();
  if (TID == 0) {
    __threadfence();
    unsigned g = __hip_atomic_load(gen, __ATOMIC_ACQUIRE, __HIP_MEMORY_SCOPE_AGENT);
    unsigned old = __hip_atomic_fetch_add(cnt, 1u, __ATOMIC_ACQ_REL, __HIP_MEMORY_SCOPE_AGENT);
    if (old == (unsigned)(gridDim.x - 1)) {
      __hip_atomic_store(cnt, 0u, __ATOMIC_RELAXED, __HIP_MEMORY_SCOPE_AGENT);
      __hip_atomic_fetch_add(gen, 1u, __ATOMIC_RELEASE, __HIP_MEMORY_SCOPE_AGENT);
    } else {
      while (__hip_atomic_load(gen, __ATOMIC_ACQUIRE, __HIP_MEMORY_SCOPE_AGENT) == g)
        __builtin_amdgcn_s_sleep(2);
    }
  }
  __syncthreads();
}

// Pack fp32 activations [4,K] into WMMA A-fragment bf16 layout (rows 4..15 = 0).
// A operand 16x32: lane m = lane&15; lanes 0-15 hold K = kc*32 + {0..7,16..23},
// lanes 16-31 hold K = kc*32 + {8..15,24..31}; 16 bf16 contiguous per lane.
__device__ void pack_A(const float* __restrict__ act, unsigned short* __restrict__ dst,
                       int K) {
  int kch = K >> 5;
  for (int idx = TID; idx < kch * 512; idx += kThreads) {
    int kc = idx >> 9, r = idx & 511, lane = r >> 4, i = r & 15;
    int m = lane & 15, kh = lane >> 4;
    int k = kc * 32 + kh * 8 + (i < 8 ? i : 8 + i);
    float v = (m < 4) ? act[m * K + k] : 0.f;
    dst[idx] = f2bf_u(v);
  }
}

// Copy packed A fragments from global to LDS (async-LDS path when available).
__device__ void stage_to_lds(const unsigned short* __restrict__ g,
                             unsigned short* __restrict__ s, int elems) {
#if defined(ASYNC_LDS_COPY)
  for (int i = TID * 8; i < elems; i += kThreads * 8)
    __builtin_amdgcn_global_load_async_to_lds_b128(
        (v4i*)(g + i), (v4i*)(s + i), 0, 0);
#if __has_builtin(__builtin_amdgcn_s_wait_asynccnt)
  __builtin_amdgcn_s_wait_asynccnt(0);
#else
  asm volatile("s_wait_asynccnt 0x0" ::: "memory");
#endif
#else
  for (int i = TID * 8; i < elems; i += kThreads * 8) {
    uint4 v = *(const uint4*)(g + i);
    *(uint4*)(s + i) = v;
  }
#endif
}

// One 16x16 output tile (rows 0..3 valid): A fragments from LDS, B from global.
__device__ v8f mm_tile_lds(const unsigned short* __restrict__ sA,
                           const unsigned short* __restrict__ Bp,
                           int kchunks, int lane) {
  v8f acc = {0.f, 0.f, 0.f, 0.f, 0.f, 0.f, 0.f, 0.f};
  const unsigned short* ap = sA + lane * 16;
  const unsigned short* bp = Bp + lane * 16;
  for (int kc = 0; kc < kchunks; ++kc) {
    __builtin_prefetch(bp + 2 * 512, 0, 1);      // global_prefetch ahead
    v16bf a = *(const v16bf*)ap;                 // ds_load_b128 x2
    v16bf b = *(const v16bf*)bp;                 // global_load_b128 x2
    ap += 512; bp += 512;
    acc = __builtin_amdgcn_wmma_f32_16x16x32_bf16(
        false, a, false, b, (short)0, acc, false, false);
  }
  return acc;
}

__device__ void ln_mix_stage(const float* __restrict__ x,
                             const float* lw, const float* lb,
                             const float* mixk, const float* mixv, const float* mixr,
                             float* prev, float* ok, float* ov, float* orr,
                             float* r1, float* r2s) {
  for (int b = 0; b < kB; ++b) {
    const float* xb = x + b * kD;
    float s = 0.f, s2 = 0.f;
    for (int d = TID; d < kD; d += kThreads) { float v = xb[d]; s += v; s2 += v * v; }
    r1[TID] = s; r2s[TID] = s2; __syncthreads();
    for (int o = kThreads / 2; o > 0; o >>= 1) {
      if (TID < o) { r1[TID] += r1[TID + o]; r2s[TID] += r2s[TID + o]; }
      __syncthreads();
    }
    float mu = r1[0] * (1.f / kD);
    float var = r2s[0] * (1.f / kD) - mu * mu;
    float rstd = rsqrtf(var + 1e-5f);
    __syncthreads();
    for (int d = TID; d < kD; d += kThreads) {
      float xn = (xb[d] - mu) * rstd * lw[d] + lb[d];
      float pv = prev[b * kD + d];
      float a = mixk[d]; ok[b * kD + d] = xn * a + pv * (1.f - a);
      if (ov) { float c = mixv[d]; ov[b * kD + d] = xn * c + pv * (1.f - c); }
      float r = mixr[d]; orr[b * kD + d] = xn * r + pv * (1.f - r);
      prev[b * kD + d] = xn;
    }
    __syncthreads();
  }
}

__device__ void wkv_stage(const RwkvParams& p, int l) {
  const int base = l * kB * kD;
  for (int i = TID; i < kB * kD; i += kThreads) {
    int d = i % kD;
    int ld = l * kD + d;
    float kv = p.kout[i], vv = p.vout[i], rv = p.rout[i];
    float a = p.aa[base + i], b = p.bb[base + i], pv = p.pp[base + i];
    float ww = p.tf[ld] + kv;
    float q = fmaxf(pv, ww);
    float e1 = __expf(pv - q), e2 = __expf(ww - q);
    float wkv = (e1 * a + e2 * vv) / (e1 * b + e2);
    float ww2 = pv + p.td[ld];
    float q2 = fmaxf(ww2, kv);
    float e1b = __expf(ww2 - q2), e2b = __expf(kv - q2);
    p.aa[base + i] = e1b * a + e2b * vv;
    p.bb[base + i] = e1b * b + e2b;
    p.pp[base + i] = q2;
    p.rg[i] = wkv / (1.f + __expf(-rv));         // sigmoid(r)*wkv
  }
}

__device__ void ffn_elem_stage(const RwkvParams& p) {
  for (int i = TID; i < kB * kD; i += kThreads)
    p.r2sig[i] = 1.f / (1.f + __expf(-p.r2raw[i]));
  for (int j = TID; j < kB * kF; j += kThreads) {
    float v = fmaxf(p.kkraw[j], 0.f);
    p.kk[j] = v * v;
  }
}

__device__ void out_stage(const RwkvParams& p, int t, float* r1, float* r2s) {
  for (int b = 0; b < kB; ++b) {
    const float* xb = p.x + b * kD;
    float s = 0.f, s2 = 0.f;
    for (int d = TID; d < kD; d += kThreads) { float v = xb[d]; s += v; s2 += v * v; }
    r1[TID] = s; r2s[TID] = s2; __syncthreads();
    for (int o = kThreads / 2; o > 0; o >>= 1) {
      if (TID < o) { r1[TID] += r1[TID + o]; r2s[TID] += r2s[TID + o]; }
      __syncthreads();
    }
    float mu = r1[0] * (1.f / kD);
    float var = r2s[0] * (1.f / kD) - mu * mu;
    float rstd = rsqrtf(var + 1e-5f);
    __syncthreads();
    for (int d = TID; d < kD; d += kThreads)
      p.out[((size_t)b * kT + t) * kD + d] = (xb[d] - mu) * rstd * p.lnow[d] + p.lnob[d];
    __syncthreads();
  }
}

__global__ void __launch_bounds__(kThreads)
rwkv_persistent(RwkvParams p) {
  __shared__ float r1[kThreads], r2s[kThreads];
  extern __shared__ unsigned short sA[];       // up to 96KB packed A fragments
  const int wg = blockIdx.x;
  const int lane = TID & 31;
  const int wave = TID >> 5;
  const int gwave = wg * kWavesPerWG + wave;
  const int nwaves = gridDim.x * kWavesPerWG;

  for (int t = 0; t < kT; ++t) {
    // S0: WG0 writes out(t-1) and loads x(t)
    if (wg == 0) {
      if (t > 0) out_stage(p, t - 1, r1, r2s);
      __syncthreads();
      for (int i = TID; i < kB * kD; i += kThreads) {
        int b = i / kD, d = i % kD;
        p.x[i] = p.x_seq[((size_t)b * kT + t) * kD + d];
      }
    }
    grid_barrier(&p.bar[0], &p.bar[1]);

    for (int l = 0; l < kL; ++l) {
      // S1: ln1 + time-mix, then pack A fragments for K/V/R
      if (wg == 0) {
        ln_mix_stage(p.x, p.ln1_w + l * kD, p.ln1_b + l * kD,
                     p.mk + l * kD, p.mv + l * kD, p.mr + l * kD,
                     p.xa + (size_t)l * kB * kD, p.xk, p.xv, p.xr, r1, r2s);
        __syncthreads();
        pack_A(p.xk, p.pk_xk, kD);
        pack_A(p.xv, p.pk_xv, kD);
        pack_A(p.xr, p.pk_xr, kD);
      }
      grid_barrier(&p.bar[0], &p.bar[1]);

      // S2: K, V, R matmuls (3 * 48 tiles); A fragments staged in LDS
      stage_to_lds(p.pk_xk, sA, kPkD);
      stage_to_lds(p.pk_xv, sA + kPkD, kPkD);
      stage_to_lds(p.pk_xr, sA + 2 * kPkD, kPkD);
      __syncthreads();
      for (int tile = gwave; tile < 3 * kNtD; tile += nwaves) {
        int which = tile / kNtD, nt = tile % kNtD;
        const unsigned short* W =
            (which == 0 ? p.Wk : which == 1 ? p.Wv : p.Wr) +
            (size_t)l * kDD + (size_t)nt * kKchD * 512;
        v8f acc = mm_tile_lds(sA + which * kPkD, W, kKchD, lane);
        float* O = which == 0 ? p.kout : which == 1 ? p.vout : p.rout;
        if (lane < 16) {
          int n = nt * 16 + lane;
#pragma unroll
          for (int rr = 0; rr < 4; ++rr) O[rr * kD + n] = acc[rr];
        }
      }
      grid_barrier(&p.bar[0], &p.bar[1]);

      // S3: WKV recurrence + sigmoid(r) gate, then pack rg fragments
      if (wg == 0) {
        wkv_stage(p, l);
        __syncthreads();
        pack_A(p.rg, p.pk_rg, kD);
      }
      grid_barrier(&p.bar[0], &p.bar[1]);

      // S4: x += (r*wkv) @ Wo (48 tiles)
      stage_to_lds(p.pk_rg, sA, kPkD);
      __syncthreads();
      for (int tile = gwave; tile < kNtD; tile += nwaves) {
        const unsigned short* W = p.Wo + (size_t)l * kDD + (size_t)tile * kKchD * 512;
        v8f acc = mm_tile_lds(sA, W, kKchD, lane);
        if (lane < 16) {
          int n = tile * 16 + lane;
#pragma unroll
          for (int rr = 0; rr < 4; ++rr) p.x[rr * kD + n] += acc[rr];
        }
      }
      grid_barrier(&p.bar[0], &p.bar[1]);

      // S4b: ln2 + channel-mix, pack xk2/xr2 fragments
      if (wg == 0) {
        ln_mix_stage(p.x, p.ln2_w + l * kD, p.ln2_b + l * kD,
                     p.fmk + l * kD, nullptr, p.fmr + l * kD,
                     p.xf + (size_t)l * kB * kD, p.xk2, nullptr, p.xr2, r1, r2s);
        __syncthreads();
        pack_A(p.xr2, p.pk_xr2, kD);
        pack_A(p.xk2, p.pk_xk2, kD);
      }
      grid_barrier(&p.bar[0], &p.bar[1]);

      // S5: r2raw = xr2@Fr (48 tiles) and kkraw = xk2@Fk (192 tiles)
      stage_to_lds(p.pk_xr2, sA, kPkD);
      stage_to_lds(p.pk_xk2, sA + kPkD, kPkD);
      __syncthreads();
      for (int tile = gwave; tile < kNtD + kNtF; tile += nwaves) {
        if (tile < kNtD) {
          const unsigned short* W = p.Fr + (size_t)l * kDD + (size_t)tile * kKchD * 512;
          v8f acc = mm_tile_lds(sA, W, kKchD, lane);
          if (lane < 16) {
            int n = tile * 16 + lane;
#pragma unroll
            for (int rr = 0; rr < 4; ++rr) p.r2raw[rr * kD + n] = acc[rr];
          }
        } else {
          int nt = tile - kNtD;
          const unsigned short* W = p.Fk + (size_t)l * kDF + (size_t)nt * kKchD * 512;
          v8f acc = mm_tile_lds(sA + kPkD, W, kKchD, lane);
          if (lane < 16) {
            int n = nt * 16 + lane;
#pragma unroll
            for (int rr = 0; rr < 4; ++rr) p.kkraw[rr * kF + n] = acc[rr];
          }
        }
      }
      grid_barrier(&p.bar[0], &p.bar[1]);

      // S6: sigmoid + squared ReLU, pack kk fragments (K=3072)
      if (wg == 0) {
        ffn_elem_stage(p);
        __syncthreads();
        pack_A(p.kk, p.pk_kk, kF);
      }
      grid_barrier(&p.bar[0], &p.bar[1]);

      // S7: x += r2sig * (kk @ Fv)   (48 tiles, K=3072)
      stage_to_lds(p.pk_kk, sA, kPkF);
      __syncthreads();
      for (int tile = gwave; tile < kNtD; tile += nwaves) {
        const unsigned short* W = p.Fv + (size_t)l * kDF + (size_t)tile * kKchF * 512;
        v8f acc = mm_tile_lds(sA, W, kKchF, lane);
        if (lane < 16) {
          int n = tile * 16 + lane;
#pragma unroll
          for (int rr = 0; rr < 4; ++rr)
            p.x[rr * kD + n] += p.r2sig[rr * kD + n] * acc[rr];
        }
      }
      grid_barrier(&p.bar[0], &p.bar[1]);
    }
  }
  if (wg == 0) out_stage(p, kT - 1, r1, r2s);
}

// ---------------- host ------------------------------------------------
extern "C" void kernel_launch(void* const* d_in, const int* in_sizes, int n_in,
                              void* d_out, int out_size, void* d_ws, size_t ws_size,
                              hipStream_t stream) {
  (void)in_sizes; (void)n_in; (void)out_size;
  const int*   tokens  = (const int*)  d_in[0];
  const float* emb     = (const float*)d_in[1];
  const float* ln0_w   = (const float*)d_in[2];
  const float* ln0_b   = (const float*)d_in[3];
  const float* ln1_w   = (const float*)d_in[4];
  const float* ln1_b   = (const float*)d_in[5];
  const float* ln2_w   = (const float*)d_in[6];
  const float* ln2_b   = (const float*)d_in[7];
  const float* mix_k   = (const float*)d_in[8];
  const float* mix_v   = (const float*)d_in[9];
  const float* mix_r   = (const float*)d_in[10];
  const float* tfirst  = (const float*)d_in[11];
  const float* tdecay  = (const float*)d_in[12];
  const float* att_Wk  = (const float*)d_in[13];
  const float* att_Wv  = (const float*)d_in[14];
  const float* att_Wr  = (const float*)d_in[15];
  const float* att_Wo  = (const float*)d_in[16];
  const float* fmix_k  = (const float*)d_in[17];
  const float* fmix_r  = (const float*)d_in[18];
  const float* ffn_Wk  = (const float*)d_in[19];
  const float* ffn_Wv  = (const float*)d_in[20];
  const float* ffn_Wr  = (const float*)d_in[21];
  const float* lnout_w = (const float*)d_in[22];
  const float* lnout_b = (const float*)d_in[23];

  char* ws = (char*)d_ws;
  size_t off = 0;
  auto take = [&](size_t bytes) -> void* {
    void* p = ws + off;
    off += (bytes + 255) & ~(size_t)255;
    return p;
  };
  unsigned short* Wk = (unsigned short*)take((size_t)kL * kDD * 2);
  unsigned short* Wv = (unsigned short*)take((size_t)kL * kDD * 2);
  unsigned short* Wr = (unsigned short*)take((size_t)kL * kDD * 2);
  unsigned short* Wo = (unsigned short*)take((size_t)kL * kDD * 2);
  unsigned short* Fr = (unsigned short*)take((size_t)kL * kDD * 2);
  unsigned short* Fk = (unsigned short*)take((size_t)kL * kDF * 2);
  unsigned short* Fv = (unsigned short*)take((size_t)kL * kDF * 2);
  float* x_seq = (float*)take((size_t)kB * kT * kD * 4);
  float* x     = (float*)take((size_t)kB * kD * 4);
  float* xa    = (float*)take((size_t)kL * kB * kD * 4);
  float* xf    = (float*)take((size_t)kL * kB * kD * 4);
  float* aa    = (float*)take((size_t)kL * kB * kD * 4);
  float* bb    = (float*)take((size_t)kL * kB * kD * 4);
  float* pp    = (float*)take((size_t)kL * kB * kD * 4);
  float* xk    = (float*)take((size_t)kB * kD * 4);
  float* xv    = (float*)take((size_t)kB * kD * 4);
  float* xr    = (float*)take((size_t)kB * kD * 4);
  float* kout  = (float*)take((size_t)kB * kD * 4);
  float* vout  = (float*)take((size_t)kB * kD * 4);
  float* rout  = (float*)take((size_t)kB * kD * 4);
  float* rg    = (float*)take((size_t)kB * kD * 4);
  float* xk2   = (float*)take((size_t)kB * kD * 4);
  float* xr2   = (float*)take((size_t)kB * kD * 4);
  float* r2raw = (float*)take((size_t)kB * kD * 4);
  float* r2sig = (float*)take((size_t)kB * kD * 4);
  float* kkraw = (float*)take((size_t)kB * kF * 4);
  float* kkbuf = (float*)take((size_t)kB * kF * 4);
  unsigned short* pk_xk  = (unsigned short*)take((size_t)kPkD * 2);
  unsigned short* pk_xv  = (unsigned short*)take((size_t)kPkD * 2);
  unsigned short* pk_xr  = (unsigned short*)take((size_t)kPkD * 2);
  unsigned short* pk_rg  = (unsigned short*)take((size_t)kPkD * 2);
  unsigned short* pk_xk2 = (unsigned short*)take((size_t)kPkD * 2);
  unsigned short* pk_xr2 = (unsigned short*)take((size_t)kPkD * 2);
  unsigned short* pk_kk  = (unsigned short*)take((size_t)kPkF * 2);
  unsigned* bar = (unsigned*)take(256);
  if (off > ws_size) return;  // workspace too small; nothing safe to do

  struct PackJob { const float* src; unsigned short* dst; int K, N; };
  PackJob jobs[7] = {
    {att_Wk, Wk, kD, kD}, {att_Wv, Wv, kD, kD}, {att_Wr, Wr, kD, kD},
    {att_Wo, Wo, kD, kD}, {ffn_Wr, Fr, kD, kD},
    {ffn_Wk, Fk, kD, kF}, {ffn_Wv, Fv, kF, kD},
  };
  for (int j = 0; j < 7; ++j) {
    int total = kL * jobs[j].K * jobs[j].N / 16;
    pack_weights<<<(total + kThreads - 1) / kThreads, kThreads, 0, stream>>>(
        jobs[j].src, jobs[j].dst, jobs[j].K, jobs[j].N, total);
  }
  int n_state = kL * kB * kD;
  init_state<<<(n_state + kThreads - 1) / kThreads, kThreads, 0, stream>>>(
      xa, xf, aa, bb, pp, bar);
  embed_ln0<<<kB * kT, kThreads, 0, stream>>>(tokens, emb, ln0_w, ln0_b, x_seq);

  RwkvParams P;
  P.Wk = Wk; P.Wv = Wv; P.Wr = Wr; P.Wo = Wo; P.Fr = Fr; P.Fk = Fk; P.Fv = Fv;
  P.ln1_w = ln1_w; P.ln1_b = ln1_b; P.ln2_w = ln2_w; P.ln2_b = ln2_b;
  P.mk = mix_k; P.mv = mix_v; P.mr = mix_r; P.tf = tfirst; P.td = tdecay;
  P.fmk = fmix_k; P.fmr = fmix_r; P.lnow = lnout_w; P.lnob = lnout_b;
  P.x_seq = x_seq; P.x = x; P.xa = xa; P.xf = xf; P.aa = aa; P.bb = bb; P.pp = pp;
  P.xk = xk; P.xv = xv; P.xr = xr; P.kout = kout; P.vout = vout; P.rout = rout;
  P.rg = rg; P.xk2 = xk2; P.xr2 = xr2; P.r2raw = r2raw; P.kkraw = kkraw;
  P.kk = kkbuf; P.r2sig = r2sig;
  P.pk_xk = pk_xk; P.pk_xv = pk_xv; P.pk_xr = pk_xr; P.pk_rg = pk_rg;
  P.pk_xk2 = pk_xk2; P.pk_xr2 = pk_xr2; P.pk_kk = pk_kk;
  P.out = (float*)d_out; P.bar = bar;

  size_t dynShmem = (size_t)kPkF * 2;          // 96 KB packed-A staging
  rwkv_persistent<<<kNWG, kThreads, dynShmem, stream>>>(P);
}
